// AutoEncoder_29918742184170
// MI455X (gfx1250) — matrix operand order repaired and checked by reading, not compiled, not run on hardware
//
#include <hip/hip_runtime.h>
#include <hip/hip_bf16.h>
#include <stdint.h>

// ---------------------------------------------------------------------------
// RBF autoencoder forward, fused for MI455X (gfx1250, wave32, WMMA bf16).
//   stage1: z  = exp(x.cencT - bias) @ alpha_encT      (105 GFLOP GEMM fused)
//   stage2: out= exp(z.cdecT - bias) @ alpha_dec       (105 GFLOP GEMM fused)
// exp(-||a-b||^2/2) = exp(a.b - 0.5||a||^2 - 0.5||b||^2); norms kept in f32.
// ---------------------------------------------------------------------------

typedef __attribute__((ext_vector_type(16))) __bf16  v16bf;
typedef __attribute__((ext_vector_type(8)))  float   v8f;
typedef __attribute__((ext_vector_type(8)))  unsigned int v8u;

#define NROW 8192   // dataset size / #centers
#define FDIM 784    // feature dim
#define FPAD 800    // feature dim padded to multiple of 32 (zero fill)
#define LDIM 20     // latent dim
#define LPAD 32     // latent dim padded to 32 (zero fill)

__device__ __forceinline__ unsigned short f2bf(float f) {
  unsigned int u = __float_as_uint(f);
  u += 0x7FFFu + ((u >> 16) & 1u);          // round-to-nearest-even
  return (unsigned short)(u >> 16);
}

// load a 16-element bf16 fragment as two 16B vectors (addresses 16B aligned)
__device__ __forceinline__ v8u ld_frag(const unsigned short* p0,
                                       const unsigned short* p1) {
  uint4 a = *reinterpret_cast<const uint4*>(p0);
  uint4 b = *reinterpret_cast<const uint4*>(p1);
  v8u r = {a.x, a.y, a.z, a.w, b.x, b.y, b.z, b.w};
  return r;
}

__device__ __forceinline__ v8f wmma_bf16f32(v8u a, v8u b, v8f c) {
  return __builtin_amdgcn_wmma_f32_16x16x32_bf16(
      false, __builtin_bit_cast(v16bf, a),
      false, __builtin_bit_cast(v16bf, b),
      (short)0, c, false, false);
}

// --------------------------- prep kernels ----------------------------------

// dst[Rp x Cp] bf16 = zero-padded convert of src[R x C] f32 (total = Rp*Cp)
__global__ void k_pad_bf16(const float* __restrict__ src,
                           unsigned short* __restrict__ dst,
                           int R, int C, int Cp, long total) {
  long i = (long)blockIdx.x * blockDim.x + threadIdx.x;
  if (i >= total) return;
  int r = (int)(i / Cp);
  int c = (int)(i - (long)r * Cp);
  float v = (r < R && c < C) ? src[(long)r * C + c] : 0.0f;
  dst[i] = f2bf(v);
}

// alpha_dec [N x F] f32 -> adT [F x N] bf16 (so decoder B-fragments are contiguous)
__global__ void k_transpose_bf16(const float* __restrict__ src,
                                 unsigned short* __restrict__ dst) {
  long total = (long)FDIM * NROW;
  long i = (long)blockIdx.x * blockDim.x + threadIdx.x;
  if (i >= total) return;
  int f = (int)(i / NROW);
  int j = (int)(i - (long)f * NROW);
  dst[i] = f2bf(src[(long)j * FDIM + f]);
}

// dst[r] = 0.5 * sum_c src[r][c]^2   (one block per row)
__global__ void k_halfnorm(const float* __restrict__ src,
                           float* __restrict__ dst, int C) {
  __shared__ float red[256];
  int r = blockIdx.x;
  float s = 0.f;
  for (int c = threadIdx.x; c < C; c += 256) {
    float v = src[(long)r * C + c];
    s += v * v;
  }
  red[threadIdx.x] = s;
  __syncthreads();
  for (int o = 128; o > 0; o >>= 1) {
    if (threadIdx.x < o) red[threadIdx.x] += red[threadIdx.x + o];
    __syncthreads();
  }
  if (threadIdx.x == 0) dst[r] = 0.5f * red[0];
}

// --------------------------- fused encoder ---------------------------------
// Workgroup: 16 rows of x, 8 waves split the 8192 centers in 32-col chunks.
// Per chunk/wave: 50 bf16 WMMAs (K=800 distance GEMM, two 16-col subtiles),
// exp bias, LDS-transpose of the P tile, 2 WMMAs vs padded alpha_enc -> zacc.
__global__ __launch_bounds__(256)
void k_encoder(const unsigned short* __restrict__ xbf,   // [8192 x 800]
               const unsigned short* __restrict__ cbf,   // [8192 x 800]
               const unsigned short* __restrict__ aebf,  // [32 x 8192], rows>=20 zero
               const float* __restrict__ xs,             // 0.5*||x_i||^2
               const float* __restrict__ cs,             // 0.5*||c_j||^2
               float* __restrict__ z) {                  // [8192 x 20] f32
  __shared__ __align__(16) unsigned short pbuf[8][16 * 32];
  __shared__ float zred[8][16][32];
  const int tid  = threadIdx.x;
  const int w    = tid >> 5;
  const int lane = tid & 31;
  const int n    = lane & 15;
  const int hi   = lane >> 4;          // lane half
  const int kofsA = hi ? 16 : 0;       // A-frag K offset (elements)
  const int kofsB = hi ? 16 : 0;       // B-frag K offset (elements)
  const int row0 = blockIdx.x * 16;

  float xsh[8];
#pragma unroll
  for (int r = 0; r < 8; ++r) xsh[r] = xs[row0 + r + 8 * hi];

  const unsigned short* xrow = xbf + (long)(row0 + n) * FPAD;

  v8f zacc0 = {}, zacc1 = {};

  for (int it = 0; it < 32; ++it) {
    const int j0 = (it * 8 + w) * 32;
    const unsigned short* crow0 = cbf + (long)(j0 + n) * FPAD + kofsB;
    const unsigned short* crow1 = cbf + (long)(j0 + 16 + n) * FPAD + kofsB;
    v8f c0 = {}, c1 = {};
    for (int kb = 0; kb < FPAD / 32; ++kb) {   // 25 K-steps
      const unsigned short* ap = xrow + kb * 32 + kofsA;
      v8u af = ld_frag(ap, ap + 8);
      v8u b0 = ld_frag(crow0 + kb * 32, crow0 + kb * 32 + 8);
      v8u b1 = ld_frag(crow1 + kb * 32, crow1 + kb * 32 + 8);
      c0 = wmma_bf16f32(af, b0, c0);
      c1 = wmma_bf16f32(af, b1, c1);
    }
    // P = exp(dot - 0.5||x||^2 - 0.5||c||^2), staged to LDS in A-layout order
    const float cs0 = cs[j0 + n];
    const float cs1 = cs[j0 + 16 + n];
#pragma unroll
    for (int r = 0; r < 8; ++r) {
      const int m = r + 8 * hi;
      pbuf[w][m * 32 + n]      = f2bf(__expf(c0[r] - xsh[r] - cs0));
      pbuf[w][m * 32 + 16 + n] = f2bf(__expf(c1[r] - xsh[r] - cs1));
    }
    // same-wave LDS ops are in-order; reload as A fragment [16m x 32j]
    const unsigned short* pp = &pbuf[w][n * 32 + kofsA];
    v8u pf = ld_frag(pp, pp + 8);
    const unsigned short* a0 = aebf + (long)n * NROW + j0 + kofsB;        // l = n
    const unsigned short* a1 = aebf + (long)(16 + n) * NROW + j0 + kofsB; // l = 16+n (pad)
    v8u be0 = ld_frag(a0, a0 + 8);
    v8u be1 = ld_frag(a1, a1 + 8);
    zacc0 = wmma_bf16f32(pf, be0, zacc0);
    zacc1 = wmma_bf16f32(pf, be1, zacc1);
  }

  // cross-wave reduction of z tiles
#pragma unroll
  for (int r = 0; r < 8; ++r) {
    const int m = r + 8 * hi;
    zred[w][m][n]      = zacc0[r];
    zred[w][m][16 + n] = zacc1[r];
  }
  __syncthreads();
  for (int p = tid; p < 16 * 32; p += 256) {
    const int m = p >> 5, l = p & 31;
    float s = 0.f;
#pragma unroll
    for (int ww = 0; ww < 8; ++ww) s += zred[ww][m][l];
    if (l < LDIM) z[(long)(row0 + m) * LDIM + l] = s;
  }
}

// --------------------------- fused decoder ---------------------------------
// Workgroup: 16 rows x 112 output cols (7 WMMA n-tiles); 8 waves split j.
// Per chunk/wave: 2 bf16 dist WMMAs (K=32 padded latent), exp bias, LDS P
// transpose, 7 WMMAs vs alpha_decT (L2-resident). Cross-wave LDS reduction.
__global__ __launch_bounds__(256)
void k_decoder(const unsigned short* __restrict__ zbf,   // [8192 x 32]
               const unsigned short* __restrict__ cdbf,  // [8192 x 32]
               const unsigned short* __restrict__ adT,   // [784 x 8192]
               const float* __restrict__ zs,             // 0.5*||z_i||^2
               const float* __restrict__ cds,            // 0.5*||cd_j||^2
               float* __restrict__ out) {                // [8192 x 784] f32
  __shared__ __align__(16) unsigned short pbuf[8][16 * 32];
  __shared__ float ored[8][16][16];
  const int tid  = threadIdx.x;
  const int w    = tid >> 5;
  const int lane = tid & 31;
  const int n    = lane & 15;
  const int hi   = lane >> 4;
  const int kofsA = hi ? 16 : 0;
  const int kofsB = hi ? 16 : 0;
  const int row0 = blockIdx.x * 16;
  const int f0   = blockIdx.y * 112;

  // A fragment of z (constant over the whole j loop)
  const unsigned short* zp = zbf + (long)(row0 + n) * LPAD + kofsA;
  v8u zf = ld_frag(zp, zp + 8);

  float zsh[8];
#pragma unroll
  for (int r = 0; r < 8; ++r) zsh[r] = zs[row0 + r + 8 * hi];

  v8f zero = {};
  v8f oacc[7];
#pragma unroll
  for (int t = 0; t < 7; ++t) oacc[t] = zero;

  for (int it = 0; it < 32; ++it) {
    const int j0 = (it * 8 + w) * 32;
    const unsigned short* c0p = cdbf + (long)(j0 + n) * LPAD + kofsB;
    const unsigned short* c1p = cdbf + (long)(j0 + 16 + n) * LPAD + kofsB;
    v8f p0 = wmma_bf16f32(zf, ld_frag(c0p, c0p + 8), zero);
    v8f p1 = wmma_bf16f32(zf, ld_frag(c1p, c1p + 8), zero);
    const float cd0 = cds[j0 + n];
    const float cd1 = cds[j0 + 16 + n];
#pragma unroll
    for (int r = 0; r < 8; ++r) {
      const int m = r + 8 * hi;
      pbuf[w][m * 32 + n]      = f2bf(__expf(p0[r] - zsh[r] - cd0));
      pbuf[w][m * 32 + 16 + n] = f2bf(__expf(p1[r] - zsh[r] - cd1));
    }
    const unsigned short* pp = &pbuf[w][n * 32 + kofsA];
    v8u pf = ld_frag(pp, pp + 8);
#pragma unroll
    for (int t = 0; t < 7; ++t) {
      const unsigned short* bp = adT + (long)(f0 + t * 16 + n) * NROW + j0 + kofsB;
      oacc[t] = wmma_bf16f32(pf, ld_frag(bp, bp + 8), oacc[t]);
    }
  }

  // cross-wave reduction, one 16x16 subtile at a time (256 threads = 256 elems)
  for (int t = 0; t < 7; ++t) {
#pragma unroll
    for (int r = 0; r < 8; ++r) ored[w][r + 8 * hi][n] = oacc[t][r];
    __syncthreads();
    {
      const int m = tid >> 4, nn = tid & 15;
      float s = 0.f;
#pragma unroll
      for (int ww = 0; ww < 8; ++ww) s += ored[ww][m][nn];
      out[(long)(row0 + m) * FDIM + f0 + t * 16 + nn] = s;
    }
    __syncthreads();
  }
}

// --------------------------- launcher --------------------------------------

extern "C" void kernel_launch(void* const* d_in, const int* in_sizes, int n_in,
                              void* d_out, int out_size, void* d_ws, size_t ws_size,
                              hipStream_t stream) {
  const float* x    = (const float*)d_in[0];   // [8192 x 784]
  const float* cenc = (const float*)d_in[1];   // [8192 x 784]
  const float* cdec = (const float*)d_in[2];   // [8192 x 20]
  const float* aenc = (const float*)d_in[3];   // [20 x 8192]
  const float* adec = (const float*)d_in[4];   // [8192 x 784]
  float* out = (float*)d_out;                  // [8192 x 784]

  char* p = (char*)d_ws;
  auto carve = [&](size_t bytes) {
    char* q = p;
    p += (bytes + 255) & ~(size_t)255;
    return q;
  };
  unsigned short* xbf  = (unsigned short*)carve((size_t)NROW * FPAD * 2);
  unsigned short* cbf  = (unsigned short*)carve((size_t)NROW * FPAD * 2);
  unsigned short* aebf = (unsigned short*)carve((size_t)LPAD * NROW * 2);
  unsigned short* adT  = (unsigned short*)carve((size_t)FDIM * NROW * 2);
  unsigned short* cdbf = (unsigned short*)carve((size_t)NROW * LPAD * 2);
  unsigned short* zbf  = (unsigned short*)carve((size_t)NROW * LPAD * 2);
  float* xs  = (float*)carve((size_t)NROW * 4);
  float* cs  = (float*)carve((size_t)NROW * 4);
  float* cds = (float*)carve((size_t)NROW * 4);
  float* zs  = (float*)carve((size_t)NROW * 4);
  float* z   = (float*)carve((size_t)NROW * LDIM * 4);

  const int T = 256;
  long t1 = (long)NROW * FPAD;
  k_pad_bf16<<<(t1 + T - 1) / T, T, 0, stream>>>(x, xbf, NROW, FDIM, FPAD, t1);
  k_pad_bf16<<<(t1 + T - 1) / T, T, 0, stream>>>(cenc, cbf, NROW, FDIM, FPAD, t1);
  long t2 = (long)LPAD * NROW;
  k_pad_bf16<<<(t2 + T - 1) / T, T, 0, stream>>>(aenc, aebf, LDIM, NROW, NROW, t2);
  long t3 = (long)FDIM * NROW;
  k_transpose_bf16<<<(t3 + T - 1) / T, T, 0, stream>>>(adec, adT);
  long t4 = (long)NROW * LPAD;
  k_pad_bf16<<<(t4 + T - 1) / T, T, 0, stream>>>(cdec, cdbf, NROW, LDIM, LPAD, t4);
  k_halfnorm<<<NROW, T, 0, stream>>>(x, xs, FDIM);
  k_halfnorm<<<NROW, T, 0, stream>>>(cenc, cs, FDIM);
  k_halfnorm<<<NROW, T, 0, stream>>>(cdec, cds, LDIM);

  k_encoder<<<NROW / 16, T, 0, stream>>>(xbf, cbf, aebf, xs, cs, z);

  k_pad_bf16<<<(t4 + T - 1) / T, T, 0, stream>>>(z, zbf, NROW, LDIM, LPAD, t4);
  k_halfnorm<<<NROW, T, 0, stream>>>(z, zs, LDIM);

  k_decoder<<<dim3(NROW / 16, FDIM / 112), T, 0, stream>>>(zbf, cdbf, adT, zs, cds, out);
}
